// GSL4SGG_56977036149414
// MI455X (gfx1250) — matrix operand order repaired and checked by reading, not compiled, not run on hardware
//
#include <hip/hip_runtime.h>
#include <hip/hip_bf16.h>

#define N_OBJ 4096
#define N_REL 32768
#define P_DIM 2048
#define D_DIM 1024
#define G_DIM 128

typedef __attribute__((ext_vector_type(16))) __bf16 v16bf;
typedef __attribute__((ext_vector_type(8)))  __bf16 v8bf;
typedef __attribute__((ext_vector_type(8)))  float  v8f;
typedef __attribute__((ext_vector_type(4)))  int    v4i;

#define BM 128
#define BN 128
#define BK 64
#define LDSS 72   // LDS row stride in bf16 elements (144B, keeps 16B alignment)

// ---- CDNA5 async global->LDS support (guarded; manual copy fallback) ------
#if defined(__has_builtin)
#  if __has_builtin(__builtin_amdgcn_global_load_async_to_lds_b128)
#    define HAVE_ASYNC 1
#  endif
#  if __has_builtin(__builtin_amdgcn_s_wait_asynccnt)
#    define HAVE_WAIT_ASYNC 1
#  endif
#endif
#ifndef HAVE_ASYNC
#  define HAVE_ASYNC 0
#endif
#ifndef HAVE_WAIT_ASYNC
#  define HAVE_WAIT_ASYNC 0
#endif

#if HAVE_ASYNC
typedef __attribute__((address_space(1))) v4i gv4i;
typedef __attribute__((address_space(3))) v4i lv4i;
#endif

__device__ __forceinline__ void async_copy_b128(const __bf16* g, __bf16* l) {
#if HAVE_ASYNC
    __builtin_amdgcn_global_load_async_to_lds_b128(
        (gv4i*)(void*)g, (lv4i*)(void*)l, 0, 0);
#else
    *(v8bf*)l = *(const v8bf*)g;
#endif
}

__device__ __forceinline__ void wait_async_zero() {
#if HAVE_ASYNC
#  if HAVE_WAIT_ASYNC
    __builtin_amdgcn_s_wait_asynccnt(0);
#  else
    asm volatile("s_wait_asynccnt 0" ::: "memory");
#  endif
#endif
}

// Assemble a 16-element bf16 WMMA fragment from two contiguous 8-element chunks.
__device__ __forceinline__ v16bf make_frag(const __bf16* p0, const __bf16* p1) {
    v8bf lo = *(const v8bf*)p0;
    v8bf hi = *(const v8bf*)p1;
    v16bf f;
#pragma unroll
    for (int i = 0; i < 8; ++i) { f[i] = lo[i]; f[i + 8] = hi[i]; }
    return f;
}

// ---------------------------------------------------------------------------
// Weight pre-pass: W[K][N] f32 -> Wt[N][K] bf16 (transposed, async-copy ready)
// ---------------------------------------------------------------------------
__global__ __launch_bounds__(256)
void convert_transpose(const float* __restrict__ W, __bf16* __restrict__ Wt,
                       int K, int N)
{
    int idx = blockIdx.x * 256 + threadIdx.x;
    if (idx < K * N) {
        int k = idx / N, n = idx % N;               // coalesced read
        Wt[(size_t)n * K + k] = (__bf16)W[idx];
    }
}

// ---------------------------------------------------------------------------
// Generic GEMM: C[M,N] = act( maybe_relu(A[M,K]) @ B[K,N] + bias[N] (+C) )
// A f32 row-major, Bt bf16 [N][K] (pre-transposed); double-buffered LDS,
// B tiles via GLOBAL_LOAD_ASYNC_TO_LDS_B128; bf16 WMMA, f32 accumulation.
// ---------------------------------------------------------------------------
__global__ __launch_bounds__(256)
void gemm_wmma(const float* __restrict__ A, const __bf16* __restrict__ Bt,
               const float* __restrict__ bias, float* __restrict__ C,
               int M, int N, int K, int relu_a, int relu_out, int accum)
{
    __shared__ __align__(16) __bf16 sA[2][BM * LDSS];
    __shared__ __align__(16) __bf16 sB[2][BN * LDSS];

    const int tid  = threadIdx.x;
    const int lane = tid & 31;
    const int wave = tid >> 5;
    const int half = (lane >> 4) & 1;
    const int l16  = lane & 15;
    const int wm   = wave & 3;   // M offset 32*wm
    const int wn   = wave >> 2;  // N offset 64*wn
    const int tileM = blockIdx.y * BM;
    const int tileN = blockIdx.x * BN;

    v8f acc[2][4];
#pragma unroll
    for (int i = 0; i < 2; ++i)
#pragma unroll
        for (int j = 0; j < 4; ++j) acc[i][j] = (v8f){0.f,0.f,0.f,0.f,0.f,0.f,0.f,0.f};

    auto stage = [&](int k0, int buf) {
        for (int i = 0; i < (BM * BK) / 256; ++i) {           // A: f32 -> bf16
            int e = i * 256 + tid;
            int r = e >> 6, c = e & 63;
            float v = A[(size_t)(tileM + r) * K + (k0 + c)];
            if (relu_a) v = fmaxf(v, 0.f);
            sA[buf][r * LDSS + c] = (__bf16)v;
        }
        for (int i = 0; i < (BN * BK) / (256 * 8); ++i) {     // B: async 16B/lane
            int ch = i * 256 + tid;                           // 8 chunks per row
            int n = ch >> 3, kk = (ch & 7) * 8;
            async_copy_b128(Bt + (size_t)(tileN + n) * K + k0 + kk,
                            &sB[buf][n * LDSS + kk]);
        }
    };

    auto compute = [&](int buf) {
#pragma unroll
        for (int kb = 0; kb < BK; kb += 32) {
            v16bf af[2], bfr[4];
#pragma unroll
            for (int i = 0; i < 2; ++i) {
                const __bf16* p = &sA[buf][(wm * 32 + i * 16 + l16) * LDSS + kb + half * 8];
                af[i] = make_frag(p, p + 16);
            }
#pragma unroll
            for (int j = 0; j < 4; ++j) {
                const __bf16* p = &sB[buf][(wn * 64 + j * 16 + l16) * LDSS + kb + half * 8];
                bfr[j] = make_frag(p, p + 16);
            }
#pragma unroll
            for (int i = 0; i < 2; ++i)
#pragma unroll
                for (int j = 0; j < 4; ++j)
                    acc[i][j] = __builtin_amdgcn_wmma_f32_16x16x32_bf16(
                        false, af[i], false, bfr[j], (short)0, acc[i][j], false, false);
        }
    };

    stage(0, 0);
    wait_async_zero();
    __syncthreads();
    int cur = 0;
    for (int k0 = 0; k0 < K; k0 += BK) {
        if (k0 + BK < K) stage(k0 + BK, cur ^ 1);   // prefetch next while computing
        compute(cur);
        wait_async_zero();
        __syncthreads();
        cur ^= 1;
    }

#pragma unroll
    for (int i = 0; i < 2; ++i) {
        int mbase = tileM + wm * 32 + i * 16 + half * 8;
#pragma unroll
        for (int j = 0; j < 4; ++j) {
            int col = tileN + wn * 64 + j * 16 + l16;
            float bv = bias ? bias[col] : 0.f;
#pragma unroll
            for (int v = 0; v < 8; ++v) {
                size_t off = (size_t)(mbase + v) * N + col;
                float c = acc[i][j][v] + bv;
                if (accum) c += C[off];
                if (relu_out) c = fmaxf(c, 0.f);
                C[off] = c;
            }
        }
    }
}

// ---------------------------------------------------------------------------
// LN stats per relation over concat(inst[pair[2r+which]], rel[r]) (2048 vals)
// ---------------------------------------------------------------------------
__global__ __launch_bounds__(256)
void ln_stats(const float* __restrict__ inst, const float* __restrict__ rel,
              const int* __restrict__ pair, int which,
              float* __restrict__ mean, float* __restrict__ rstd)
{
    __shared__ float ss[256], sq[256];
    int r = blockIdx.x, tid = threadIdx.x;
    int idx = pair[2 * r + which];
    const float* a = inst + (size_t)idx * D_DIM;
    const float* b = rel  + (size_t)r   * D_DIM;
    float s = 0.f, q = 0.f;
    for (int j = tid; j < D_DIM; j += 256) { float v = a[j]; s += v; q += v * v; }
    for (int j = tid; j < D_DIM; j += 256) { float v = b[j]; s += v; q += v * v; }
    ss[tid] = s; sq[tid] = q;
    __syncthreads();
    for (int st = 128; st > 0; st >>= 1) {
        if (tid < st) { ss[tid] += ss[tid + st]; sq[tid] += sq[tid + st]; }
        __syncthreads();
    }
    if (tid == 0) {
        float m = ss[0] * (1.f / P_DIM);
        float var = sq[0] * (1.f / P_DIM) - m * m;
        mean[r] = m;
        rstd[r] = rsqrtf(var + 1e-5f);
    }
}

// ---------------------------------------------------------------------------
// Gate: g[r] = mean_n sigmoid( relu(LN(concat(tgt,src)))[r] @ W + b ) * attn[r]
// A rows built on the fly (gather + LN + relu -> bf16); Wt bf16 [G][K] async.
// sel: -1 = identity row r, 0 = pair[2r], 1 = pair[2r+1]
// ---------------------------------------------------------------------------
__global__ __launch_bounds__(256)
void gate_wmma(const float* __restrict__ tgt, const float* __restrict__ src,
               const int* __restrict__ pair, int tgt_sel, int src_sel,
               const float* __restrict__ mean, const float* __restrict__ rstd,
               const float* __restrict__ ln_s, const float* __restrict__ ln_b,
               const __bf16* __restrict__ Wt, const float* __restrict__ bias,
               const float* __restrict__ attn, float* __restrict__ g_out)
{
    __shared__ __align__(16) __bf16 sA[2][BM * LDSS];
    __shared__ __align__(16) __bf16 sB[2][BN * LDSS];
    __shared__ float red[BM];

    const int tid  = threadIdx.x;
    const int lane = tid & 31;
    const int wave = tid >> 5;
    const int half = (lane >> 4) & 1;
    const int l16  = lane & 15;
    const int wm   = wave & 3;
    const int wn   = wave >> 2;
    const int tileM = blockIdx.x * BM;

    v8f acc[2][4];
#pragma unroll
    for (int i = 0; i < 2; ++i)
#pragma unroll
        for (int j = 0; j < 4; ++j) acc[i][j] = (v8f){0.f,0.f,0.f,0.f,0.f,0.f,0.f,0.f};

    auto stage = [&](int k0, int buf) {
        for (int i = 0; i < (BM * BK) / 256; ++i) {
            int e = i * 256 + tid;
            int r = e >> 6, c = e & 63;
            int rg = tileM + r;
            int k  = k0 + c;
            float v;
            if (k < D_DIM) {
                int rr = (tgt_sel < 0) ? rg : pair[2 * rg + tgt_sel];
                v = tgt[(size_t)rr * D_DIM + k];
            } else {
                int rr = (src_sel < 0) ? rg : pair[2 * rg + src_sel];
                v = src[(size_t)rr * D_DIM + (k - D_DIM)];
            }
            v = (v - mean[rg]) * rstd[rg] * ln_s[k] + ln_b[k];
            v = fmaxf(v, 0.f);
            sA[buf][r * LDSS + c] = (__bf16)v;
        }
        for (int i = 0; i < (G_DIM * BK) / (256 * 8); ++i) {   // 128 rows x 8 chunks
            int ch = i * 256 + tid;
            int n = ch >> 3, kk = (ch & 7) * 8;
            async_copy_b128(Wt + (size_t)n * P_DIM + k0 + kk,
                            &sB[buf][n * LDSS + kk]);
        }
    };

    auto compute = [&](int buf) {
#pragma unroll
        for (int kb = 0; kb < BK; kb += 32) {
            v16bf af[2], bfr[4];
#pragma unroll
            for (int i = 0; i < 2; ++i) {
                const __bf16* p = &sA[buf][(wm * 32 + i * 16 + l16) * LDSS + kb + half * 8];
                af[i] = make_frag(p, p + 16);
            }
#pragma unroll
            for (int j = 0; j < 4; ++j) {
                const __bf16* p = &sB[buf][(wn * 64 + j * 16 + l16) * LDSS + kb + half * 8];
                bfr[j] = make_frag(p, p + 16);
            }
#pragma unroll
            for (int i = 0; i < 2; ++i)
#pragma unroll
                for (int j = 0; j < 4; ++j)
                    acc[i][j] = __builtin_amdgcn_wmma_f32_16x16x32_bf16(
                        false, af[i], false, bfr[j], (short)0, acc[i][j], false, false);
        }
    };

    stage(0, 0);
    wait_async_zero();
    __syncthreads();
    int cur = 0;
    for (int k0 = 0; k0 < P_DIM; k0 += BK) {
        if (k0 + BK < P_DIM) stage(k0 + BK, cur ^ 1);
        compute(cur);
        wait_async_zero();
        __syncthreads();
        cur ^= 1;
    }

    // epilogue: sigmoid + mean over N=128 per row, via LDS atomic reduce
    for (int i = tid; i < BM; i += 256) red[i] = 0.f;
    __syncthreads();
#pragma unroll
    for (int i = 0; i < 2; ++i) {
#pragma unroll
        for (int j = 0; j < 4; ++j) {
            int col = wn * 64 + j * 16 + l16;
            float bv = bias[col];
#pragma unroll
            for (int v = 0; v < 8; ++v) {
                int m = wm * 32 + i * 16 + half * 8 + v;
                float y = acc[i][j][v] + bv;
                float s = 1.f / (1.f + expf(-y));
                atomicAdd(&red[m], s);
            }
        }
    }
    __syncthreads();
    for (int i = tid; i < BM; i += 256) {
        int rg = tileM + i;
        g_out[rg] = red[i] * (1.f / G_DIM) * attn[rg];
    }
}

// ---------------------------------------------------------------------------
// Small helpers
// ---------------------------------------------------------------------------
__global__ void fill_kernel(float* __restrict__ p, float v, int n) {
    int i = blockIdx.x * blockDim.x + threadIdx.x;
    if (i < n) p[i] = v;
}

__global__ void count_kernel(const int* __restrict__ pair,
                             float* __restrict__ cnt_s, float* __restrict__ cnt_o) {
    int r = blockIdx.x * blockDim.x + threadIdx.x;
    if (r < N_REL) {
        atomicAdd(&cnt_s[pair[2 * r]], 1.f);
        atomicAdd(&cnt_o[pair[2 * r + 1]], 1.f);
    }
}

// obj_msg[s] += rel[r] * 0.5*g_p2s[r]/max(cnt_s[s],1);  same for o side
__global__ __launch_bounds__(256)
void scatter_obj(const float* __restrict__ rel, const int* __restrict__ pair,
                 const float* __restrict__ g_p2s, const float* __restrict__ g_p2o,
                 const float* __restrict__ cnt_s, const float* __restrict__ cnt_o,
                 float* __restrict__ obj_msg)
{
    int r = blockIdx.x, c = threadIdx.x;        // c: float4 chunk, D/4 = 256
    int s = pair[2 * r], o = pair[2 * r + 1];
    float as = 0.5f * g_p2s[r] / fmaxf(cnt_s[s], 1.f);
    float ao = 0.5f * g_p2o[r] / fmaxf(cnt_o[o], 1.f);
    const float4* rp = (const float4*)(rel + (size_t)r * D_DIM);
    float4 v = rp[c];
    float* ps = obj_msg + (size_t)s * D_DIM + c * 4;
    float* po = obj_msg + (size_t)o * D_DIM + c * 4;
    atomicAdd(ps + 0, v.x * as); atomicAdd(ps + 1, v.y * as);
    atomicAdd(ps + 2, v.z * as); atomicAdd(ps + 3, v.w * as);
    atomicAdd(po + 0, v.x * ao); atomicAdd(po + 1, v.y * ao);
    atomicAdd(po + 2, v.z * ao); atomicAdd(po + 3, v.w * ao);
}

// pred_msg[r] = 0.5*(inst[subj]*g_s2p[r] + inst[objn]*g_o2p[r])
__global__ __launch_bounds__(256)
void pred_msg_kernel(const float* __restrict__ inst, const int* __restrict__ pair,
                     const float* __restrict__ g_s2p, const float* __restrict__ g_o2p,
                     float* __restrict__ pred_msg)
{
    int r = blockIdx.x, c = threadIdx.x;
    int s = pair[2 * r], o = pair[2 * r + 1];
    float gs = 0.5f * g_s2p[r], go = 0.5f * g_o2p[r];
    const float4* ap = (const float4*)(inst + (size_t)s * D_DIM);
    const float4* bp = (const float4*)(inst + (size_t)o * D_DIM);
    float4 a = ap[c], b = bp[c], out;
    out.x = a.x * gs + b.x * go; out.y = a.y * gs + b.y * go;
    out.z = a.z * gs + b.z * go; out.w = a.w * gs + b.w * go;
    ((float4*)(pred_msg + (size_t)r * D_DIM))[c] = out;
}

// ---------------------------------------------------------------------------
extern "C" void kernel_launch(void* const* d_in, const int* in_sizes, int n_in,
                              void* d_out, int out_size, void* d_ws, size_t ws_size,
                              hipStream_t stream) {
    const float* inst_features = (const float*)d_in[0];
    const float* rel_union     = (const float*)d_in[1];
    const int*   pair          = (const int*)  d_in[2];
    const float* attn          = (const float*)d_in[3];
    const float* obj_down_w    = (const float*)d_in[4];
    const float* obj_down_b    = (const float*)d_in[5];
    const float* rel_down_w    = (const float*)d_in[6];
    const float* rel_down_b    = (const float*)d_in[7];
    const float *g_ln_s[4], *g_ln_b[4], *g_w[4], *g_b[4];   // p2s,p2o,s2p,o2p
    for (int gi = 0; gi < 4; ++gi) {
        g_ln_s[gi] = (const float*)d_in[8 + 4 * gi + 0];
        g_ln_b[gi] = (const float*)d_in[8 + 4 * gi + 1];
        g_w[gi]    = (const float*)d_in[8 + 4 * gi + 2];
        g_b[gi]    = (const float*)d_in[8 + 4 * gi + 3];
    }
    const float* obj_wih_w  = (const float*)d_in[24];
    const float* obj_wih_b  = (const float*)d_in[25];
    const float* obj_whh_w  = (const float*)d_in[26];
    const float* obj_whh_b  = (const float*)d_in[27];
    const float* pred_wih_w = (const float*)d_in[28];
    const float* pred_wih_b = (const float*)d_in[29];
    const float* pred_whh_w = (const float*)d_in[30];
    const float* pred_whh_b = (const float*)d_in[31];
    float* out = (float*)d_out;

    char* wp = (char*)d_ws;
    auto allocf = [&](size_t n) { float* p = (float*)wp; wp += n * sizeof(float); return p; };
    auto allocb = [&](size_t n) { __bf16* p = (__bf16*)wp; wp += n * sizeof(__bf16); return p; };
    float* inst_a   = allocf((size_t)N_OBJ * D_DIM);
    float* inst_b   = allocf((size_t)N_OBJ * D_DIM);
    float* rel_a    = allocf((size_t)N_REL * D_DIM);
    float* rel_b    = allocf((size_t)N_REL * D_DIM);
    float* obj_msg  = allocf((size_t)N_OBJ * D_DIM);
    float* pred_msg = allocf((size_t)N_REL * D_DIM);
    float* mean_s   = allocf(N_REL);
    float* rstd_s   = allocf(N_REL);
    float* mean_o   = allocf(N_REL);
    float* rstd_o   = allocf(N_REL);
    float* g_p2s    = allocf(N_REL);
    float* g_p2o    = allocf(N_REL);
    float* g_s2p    = allocf(N_REL);
    float* g_o2p    = allocf(N_REL);
    float* cnt_s    = allocf(N_OBJ);
    float* cnt_o    = allocf(N_OBJ);
    // bf16 pre-transposed weights [N][K]
    __bf16* obj_down_wt = allocb((size_t)P_DIM * D_DIM);
    __bf16* rel_down_wt = allocb((size_t)P_DIM * D_DIM);
    __bf16* gate_wt[4];
    for (int gi = 0; gi < 4; ++gi) gate_wt[gi] = allocb((size_t)P_DIM * G_DIM);
    __bf16* obj_wih_wt  = allocb((size_t)D_DIM * D_DIM);
    __bf16* obj_whh_wt  = allocb((size_t)D_DIM * D_DIM);
    __bf16* pred_wih_wt = allocb((size_t)D_DIM * D_DIM);
    __bf16* pred_whh_wt = allocb((size_t)D_DIM * D_DIM);

    dim3 blk(256);

    // Weight pre-pass: convert + transpose to bf16 (async-copy ready)
    convert_transpose<<<(P_DIM * D_DIM + 255) / 256, blk, 0, stream>>>(obj_down_w, obj_down_wt, P_DIM, D_DIM);
    convert_transpose<<<(P_DIM * D_DIM + 255) / 256, blk, 0, stream>>>(rel_down_w, rel_down_wt, P_DIM, D_DIM);
    for (int gi = 0; gi < 4; ++gi)
        convert_transpose<<<(P_DIM * G_DIM + 255) / 256, blk, 0, stream>>>(g_w[gi], gate_wt[gi], P_DIM, G_DIM);
    convert_transpose<<<(D_DIM * D_DIM + 255) / 256, blk, 0, stream>>>(obj_wih_w,  obj_wih_wt,  D_DIM, D_DIM);
    convert_transpose<<<(D_DIM * D_DIM + 255) / 256, blk, 0, stream>>>(obj_whh_w,  obj_whh_wt,  D_DIM, D_DIM);
    convert_transpose<<<(D_DIM * D_DIM + 255) / 256, blk, 0, stream>>>(pred_wih_w, pred_wih_wt, D_DIM, D_DIM);
    convert_transpose<<<(D_DIM * D_DIM + 255) / 256, blk, 0, stream>>>(pred_whh_w, pred_whh_wt, D_DIM, D_DIM);

    // Down projections (relu output)
    gemm_wmma<<<dim3(D_DIM / BN, N_OBJ / BM), blk, 0, stream>>>(
        inst_features, obj_down_wt, obj_down_b, inst_a, N_OBJ, D_DIM, P_DIM, 0, 1, 0);
    gemm_wmma<<<dim3(D_DIM / BN, N_REL / BM), blk, 0, stream>>>(
        rel_union, rel_down_wt, rel_down_b, rel_a, N_REL, D_DIM, P_DIM, 0, 1, 0);

    // Segment counts (layer-invariant)
    fill_kernel<<<(2 * N_OBJ + 255) / 256, blk, 0, stream>>>(cnt_s, 0.f, 2 * N_OBJ);
    count_kernel<<<N_REL / 256, blk, 0, stream>>>(pair, cnt_s, cnt_o);

    const float* Oi = inst_a;
    const float* Or = rel_a;
    for (int L = 0; L < 2; ++L) {
        float* Ni = (L == 0) ? inst_b : out;
        float* Nr = (L == 0) ? rel_b  : out + (size_t)N_OBJ * D_DIM;

        ln_stats<<<N_REL, blk, 0, stream>>>(Oi, Or, pair, 0, mean_s, rstd_s);
        ln_stats<<<N_REL, blk, 0, stream>>>(Oi, Or, pair, 1, mean_o, rstd_o);

        // p2s: concat(inst[subj], rel);  p2o: concat(inst[objn], rel)
        gate_wmma<<<N_REL / BM, blk, 0, stream>>>(Oi, Or, pair, 0, -1, mean_s, rstd_s,
            g_ln_s[0], g_ln_b[0], gate_wt[0], g_b[0], attn, g_p2s);
        gate_wmma<<<N_REL / BM, blk, 0, stream>>>(Oi, Or, pair, 1, -1, mean_o, rstd_o,
            g_ln_s[1], g_ln_b[1], gate_wt[1], g_b[1], attn, g_p2o);
        // s2p: concat(rel, inst[subj]);  o2p: concat(rel, inst[objn])
        gate_wmma<<<N_REL / BM, blk, 0, stream>>>(Or, Oi, pair, -1, 0, mean_s, rstd_s,
            g_ln_s[2], g_ln_b[2], gate_wt[2], g_b[2], attn, g_s2p);
        gate_wmma<<<N_REL / BM, blk, 0, stream>>>(Or, Oi, pair, -1, 1, mean_o, rstd_o,
            g_ln_s[3], g_ln_b[3], gate_wt[3], g_b[3], attn, g_o2p);

        fill_kernel<<<((size_t)N_OBJ * D_DIM + 255) / 256, blk, 0, stream>>>(
            obj_msg, 0.f, N_OBJ * D_DIM);
        scatter_obj<<<N_REL, blk, 0, stream>>>(Or, pair, g_p2s, g_p2o, cnt_s, cnt_o, obj_msg);
        pred_msg_kernel<<<N_REL, blk, 0, stream>>>(Oi, pair, g_s2p, g_o2p, pred_msg);

        // Fuse: Ni = relu(obj_msg)@wih + b_ih + relu(Oi)@whh + b_hh
        gemm_wmma<<<dim3(D_DIM / BN, N_OBJ / BM), blk, 0, stream>>>(
            obj_msg, obj_wih_wt, obj_wih_b, Ni, N_OBJ, D_DIM, D_DIM, 1, 0, 0);
        gemm_wmma<<<dim3(D_DIM / BN, N_OBJ / BM), blk, 0, stream>>>(
            Oi, obj_whh_wt, obj_whh_b, Ni, N_OBJ, D_DIM, D_DIM, 1, 0, 1);
        gemm_wmma<<<dim3(D_DIM / BN, N_REL / BM), blk, 0, stream>>>(
            pred_msg, pred_wih_wt, pred_wih_b, Nr, N_REL, D_DIM, D_DIM, 1, 0, 0);
        gemm_wmma<<<dim3(D_DIM / BN, N_REL / BM), blk, 0, stream>>>(
            Or, pred_whh_wt, pred_whh_b, Nr, N_REL, D_DIM, D_DIM, 1, 0, 1);

        Oi = Ni;
        Or = Nr;
    }
}
